// Inception_57312043598390
// MI455X (gfx1250) — matrix operand order repaired and checked by reading, not compiled, not run on hardware
//
#include <hip/hip_runtime.h>
#include <hip/hip_bf16.h>

typedef __attribute__((ext_vector_type(16))) _Float16 v16h;
typedef __attribute__((ext_vector_type(8)))  _Float16 v8h;
typedef __attribute__((ext_vector_type(8)))  float    v8f;

#define B_N   32
#define CIN   64
#define LIN   8192
#define LOUT  8186
#define TT    128          // output timesteps per workgroup
#define XROWS 168          // x halo rows staged in LDS
#define XSTR  72           // x row stride (halves): 144B, 16B aligned, conflict-free
#define M1ROWS 160
#define M1STR  40          // 80B rows, 16B aligned, conflict-free
#define M2ROWS 144
#define M2STR  40
#define W1R    32
#define W1STR  200         // 400B rows

__device__ __forceinline__ v16h cat16(v8h a, v8h b) {
  return __builtin_shufflevector(a, b, 0,1,2,3,4,5,6,7,8,9,10,11,12,13,14,15);
}

__global__ __launch_bounds__(256) void incept_fused(
    const float* __restrict__ x,
    const float* __restrict__ w7_1,  const float* __restrict__ w7_3,
    const float* __restrict__ w10_1, const float* __restrict__ w10_3, const float* __restrict__ w10_5,
    const float* __restrict__ w15_1, const float* __restrict__ w15_3, const float* __restrict__ w15_5,
    float* __restrict__ out)
{
  __shared__ _Float16 xs [XROWS][XSTR];   // x tile, f16, [local_t][c]
  __shared__ _Float16 m1 [M1ROWS][M1STR]; // stage-A out: cols 0-7 b7, 8-15 b10, 16-23 b15
  __shared__ _Float16 m2 [M2ROWS][M2STR]; // stage-B out: cols 0-7 b10, 8-15 b15, 24-39 trash
  __shared__ _Float16 w1s[W1R][W1STR];    // fused first-conv weights, K' = tap*64 + c

  const int tid  = threadIdx.x;
  const int wave = tid >> 5;
  const int lane = tid & 31;
  const int n    = lane & 15;   // N index within tile (time)
  const int hi   = lane >> 4;   // half-wave
  const int t0   = blockIdx.x * TT;
  const int b    = blockIdx.y;
  const float* xb = x + (size_t)b * CIN * LIN;

  // ---- A-fragments for stages B & C, kept in registers (loaded once) ----
  // f16 A order used here: element j of lane(l) holds K = 16*hi + j
  // with our K' = tap*8 + c  =>  tap = 2*hi+(j>>3), c = j&7
  v16h aB[2];
  {
    const float* wsrc[2] = { w10_3, w15_3 };          // (6, 18), W[o, c*3+tap]
    for (int bb = 0; bb < 2; ++bb) {
      v16h a;
      for (int j = 0; j < 16; ++j) {
        int tap = 2*hi + (j>>3), c = j & 7;
        float v = (n < 6 && c < 6 && tap < 3) ? wsrc[bb][n*18 + c*3 + tap] : 0.f;
        a[j] = (_Float16)v;
      }
      aB[bb] = a;
    }
  }
  v16h aC[3];
  {
    const float* wsrc[3] = { w7_3, w10_5, w15_5 };    // (128, 6*kw)
    const int    kwv [3] = { 3, 2, 3 };
    for (int k = 0; k < 3; ++k) {
      v16h a;
      int row = wave*16 + n;                          // this wave owns M-tile == wave
      for (int j = 0; j < 16; ++j) {
        int tap = 2*hi + (j>>3), c = j & 7;
        float v = (c < 6 && tap < kwv[k]) ? wsrc[k][row*(6*kwv[k]) + c*kwv[k] + tap] : 0.f;
        a[j] = (_Float16)v;
      }
      aC[k] = a;
    }
  }

  // ---- cooperative LDS staging: x halo (coalesced over t) + fused W1 ----
  for (int idx = tid; idx < CIN * XROWS; idx += 256) {
    int c = idx / XROWS, r = idx - c * XROWS;
    int xt = t0 - 8 + r;
    float v = (xt >= 0 && xt < LIN) ? xb[(size_t)c * LIN + xt] : 0.f;
    xs[r][c] = (_Float16)v;
  }
  // rows: 0-5 w7_1, 8-13 w10_1, 16-21 w15_1, others 0; K' = tap*64 + c
  for (int idx = tid; idx < W1R * 192; idx += 256) {
    int r = idx / 192, k = idx - r * 192;
    int br = r >> 3, ch = r & 7;
    int tap = k >> 6, c = k & 63;
    float v = 0.f;
    if (br < 3 && ch < 6) {
      const float* ws = (br == 0) ? w7_1 : (br == 1) ? w10_1 : w15_1;
      v = ws[ch*192 + c*3 + tap];
    }
    w1s[r][k] = (_Float16)v;
  }
  __syncthreads();

  // ---- stage A: fused 64->18 convs.  M=2 tiles x N=10 tiles x 6 k-steps ----
  for (int combo = wave; combo < 20; combo += 8) {
    int mtile = combo / 10;
    int nt    = combo - mtile * 10;
    int am    = mtile*16 + n;          // W1 row
    int brow  = nt*16 + n;             // m1 row produced by this lane's column
    v8f acc = {};
    for (int s = 0; s < 6; ++s) {
      int k0 = 32*s + 8*hi;
      v16h a = cat16(*(const v8h*)&w1s[am][k0], *(const v8h*)&w1s[am][k0 + 16]);
      int tap = s >> 1;
      int cb  = 32*(s & 1) + 8*hi;
      const _Float16* xr = &xs[brow + tap][cb];
      v16h bf = cat16(*(const v8h*)xr, *(const v8h*)(xr + 16));
      acc = __builtin_amdgcn_wmma_f32_16x16x32_f16(false, a, false, bf,
                                                   (short)0, acc, false, false);
    }
    v8h st;                            // lane holds 8 contiguous channels for time brow
    for (int r = 0; r < 8; ++r) st[r] = (_Float16)fmaxf(acc[r], 0.f);
    *(v8h*)&m1[brow][mtile*16 + 8*hi] = st;
  }
  __syncthreads();

  // ---- stage B: 6->6 d=2 convs (branches 10, 15), K=24 pad 32 ----
  for (int combo = wave; combo < 18; combo += 8) {
    int bb   = combo / 9;
    int nt   = combo - bb * 9;
    int colb = 8 * (bb + 1);           // m1 column base for this branch's input
    int rb   = nt*16 + n;
    v8h z = {};
    v8h b0  = *(const v8h*)&m1[rb + 4*hi][colb];   // tap = 2*hi  (0 or 2)
    v8h b1v = *(const v8h*)&m1[rb + 2   ][colb];   // tap = 1
    v16h bf = cat16(b0, hi ? z : b1v);             // tap 3 is zero padding
    v8f acc = {};
    acc = __builtin_amdgcn_wmma_f32_16x16x32_f16(false, aB[bb], false, bf,
                                                 (short)0, acc, false, false);
    v8h st;
    for (int r = 0; r < 8; ++r) st[r] = (_Float16)fmaxf(acc[r], 0.f);
    *(v8h*)&m2[rb][8*bb + (hi ? 24 : 0)] = st;     // hi half writes to trash cols
  }
  __syncthreads();
  // causal zeros at t<0 come for free: x pad=0 -> m1=0 -> m2=0 (no biases).

  // ---- stage C: three 128-ch output GEMMs, this wave owns M-tile `wave` ----
  for (int k = 0; k < 3; ++k) {
    for (int nt = 0; nt < TT/16; ++nt) {
      int lt = nt*16 + n + 8;          // local row of out time t = t0 + nt*16 + n
      v8h z = {};
      v16h bf;
      if (k == 0) {                    // b7: m1 cols 0-7, taps at t, t+2, t+4
        v8h b0  = *(const v8h*)&m1[lt + 4*hi][0];
        v8h b1v = *(const v8h*)&m1[lt + 2   ][0];
        bf = cat16(b0, hi ? z : b1v);
      } else if (k == 1) {             // b10: m2 cols 0-7, taps at t-4, t
        v8h b0v = *(const v8h*)&m2[lt - 4][0];
        v8h b1v = *(const v8h*)&m2[lt    ][0];
        bf = cat16(hi ? z : b0v, hi ? z : b1v);
      } else {                         // b15: m2 cols 8-15, taps at t-8, t-4, t
        v8h b0  = *(const v8h*)&m2[lt - 8 + 8*hi][8];
        v8h b1v = *(const v8h*)&m2[lt - 4        ][8];
        bf = cat16(b0, hi ? z : b1v);
      }
      v8f acc = {};
      acc = __builtin_amdgcn_wmma_f32_16x16x32_f16(false, aC[k], false, bf,
                                                   (short)0, acc, false, false);
      int t = t0 + nt*16 + n;
      if (t < LOUT) {
        size_t base = ((size_t)b * 384 + (size_t)(k*128 + wave*16 + 8*hi)) * LOUT + t;
        for (int r = 0; r < 8; ++r)
          out[base + (size_t)r * LOUT] = fmaxf(acc[r], 0.f);
      }
    }
  }
}

extern "C" void kernel_launch(void* const* d_in, const int* in_sizes, int n_in,
                              void* d_out, int out_size, void* d_ws, size_t ws_size,
                              hipStream_t stream) {
  (void)in_sizes; (void)n_in; (void)out_size; (void)d_ws; (void)ws_size;
  dim3 grid((LOUT + TT - 1) / TT, B_N);   // 64 x 32
  incept_fused<<<grid, 256, 0, stream>>>(
      (const float*)d_in[0],
      (const float*)d_in[1], (const float*)d_in[2],
      (const float*)d_in[3], (const float*)d_in[4], (const float*)d_in[5],
      (const float*)d_in[6], (const float*)d_in[7], (const float*)d_in[8],
      (float*)d_out);
}
